// MultiHeadTEAttention_54589034332766
// MI455X (gfx1250) — matrix operand order repaired and checked by reading, not compiled, not run on hardware
//
#include <hip/hip_runtime.h>
#include <hip/hip_bf16.h>

#define M_   8
#define NQ_  512
#define NKV_ 512
#define DX_  64
#define DY_  256
#define DH_  128

typedef __attribute__((ext_vector_type(16))) __bf16 v16bf;
typedef __attribute__((ext_vector_type(8)))  float  v8f;
typedef __attribute__((ext_vector_type(8)))  unsigned short u16x8;

union Frag {
  v16bf bf;
  u16x8 h[2];
};

__device__ __forceinline__ unsigned short f2bf(float x) {
  unsigned u = __float_as_uint(x);
  u += 0x7FFFu + ((u >> 16) & 1u);   // round-to-nearest-even
  return (unsigned short)(u >> 16);
}

__device__ __forceinline__ v8f wmma_bf16(const Frag& a, const Frag& b, v8f c) {
  return __builtin_amdgcn_wmma_f32_16x16x32_bf16(
      /*neg_a=*/false, a.bf, /*neg_b=*/false, b.bf,
      /*c_mod=*/(short)0, c, /*reuse_a=*/false, /*reuse_b=*/false);
}

// ---------------- elementwise converts / norms ----------------

__global__ void cvt_bf16(const float* __restrict__ src,
                         unsigned short* __restrict__ dst, int n) {
  int i = blockIdx.x * blockDim.x + threadIdx.x;
  if (i < n) dst[i] = f2bf(src[i]);
}

// src is rows x cols row-major; dst is cols x rows row-major (bf16)
__global__ void cvt_bf16_T(const float* __restrict__ src,
                           unsigned short* __restrict__ dst, int rows, int cols) {
  int i = blockIdx.x * blockDim.x + threadIdx.x;
  if (i < rows * cols) {
    int r = i / cols, c = i % cols;
    dst[c * rows + r] = f2bf(src[i]);
  }
}

__global__ void row_sumsq(const float* __restrict__ x, float* __restrict__ out,
                          int nrows, int d) {
  int r = blockIdx.x * blockDim.x + threadIdx.x;
  if (r < nrows) {
    const float* p = x + (size_t)r * d;
    float s = 0.f;
    for (int i = 0; i < d; ++i) s += p[i] * p[i];
    out[r] = s;
  }
}

// ---------------- V = yv @ W_v (bf16 WMMA, store transposed [m][h][kv]) ----
// 4 waves per block; wave w computes h-tiles {2w, 2w+1}.

__global__ void __launch_bounds__(128)
vproj_kernel(const unsigned short* __restrict__ Yb,    // [M][NKV][DY] bf16
             const unsigned short* __restrict__ WvT,   // [DH][DY] bf16
             unsigned short* __restrict__ VbT) {       // [M][DH][NKV] bf16
  int bid = blockIdx.x;
  int m   = bid >> 5;            // NKV/16 = 32 tiles per m
  int kv0 = (bid & 31) * 16;
  int tid = threadIdx.x;
  int w   = tid >> 5;            // wave 0..3
  int L   = tid & 31;
  int lo  = L & 15, hi = L >> 4;

  const unsigned short* arow = Yb + (size_t)(m * NKV_ + kv0 + lo) * DY_;

  v8f acc[2];
  acc[0] = v8f{};
  acc[1] = v8f{};

  for (int k0 = 0; k0 < DY_; k0 += 32) {
    Frag a;
    a.h[0] = *(const u16x8*)(arow + k0 + 16 * hi);
    a.h[1] = *(const u16x8*)(arow + k0 + 16 * hi + 8);
    for (int tt = 0; tt < 2; ++tt) {
      int t = 2 * w + tt;
      const unsigned short* bp = WvT + (size_t)(16 * t + lo) * DY_ + k0 + 16 * hi;
      Frag b;
      b.h[0] = *(const u16x8*)(bp);
      b.h[1] = *(const u16x8*)(bp + 8);
      acc[tt] = wmma_bf16(a, b, acc[tt]);
    }
  }
  for (int tt = 0; tt < 2; ++tt) {
    int t = 2 * w + tt;
    int h = 16 * t + lo;
    unsigned short* dst = VbT + (size_t)m * DH_ * NKV_ + (size_t)h * NKV_;
    for (int j = 0; j < 8; ++j) {
      int kv = kv0 + j + 8 * hi;
      dst[kv] = f2bf(acc[tt][j]);
    }
  }
}

// ---------------- fused attention: scores -> softmax -> P@V -> @W_out + b ---
// 4 waves per block. Wave w: 8 kv-tiles of scores, 2 h-tiles of P@V,
// 4 n-tiles of the output projection. Softmax reduced across waves via LDS.

__global__ void __launch_bounds__(128)
attn_kernel(const unsigned short* __restrict__ Qb,    // [M][NQ][DX] bf16
            const unsigned short* __restrict__ Kb,    // [M][NKV][DX] bf16
            const unsigned short* __restrict__ VbT,   // [M][DH][NKV] bf16
            const unsigned short* __restrict__ WoT,   // [DY][DH] bf16
            const float* __restrict__ qn,             // [M*NQ]
            const float* __restrict__ kn,             // [M*NKV]
            const unsigned char* __restrict__ mask,   // [M][NQ][NKV] bool
            const float* __restrict__ b_out,          // [DY]
            const float* __restrict__ lengthscale,    // [1]
            float* __restrict__ out) {                // [M][NQ][DY] f32
  __shared__ float          S[16 * NKV_];       // 32 KB
  __shared__ unsigned short P[16 * NKV_];       // 16 KB
  __shared__ unsigned short Ol[16 * DH_];       //  4 KB
  __shared__ float          red[128];
  __shared__ float          rowmax[16];
  __shared__ float          inv_sum[16];

  int bid = blockIdx.x;
  int m   = bid >> 5;           // NQ/16 = 32 tiles per m
  int q0  = (bid & 31) * 16;
  int tid = threadIdx.x;
  int w   = tid >> 5;           // wave 0..3
  int L   = tid & 31;
  int lo  = L & 15, hi = L >> 4;

  float l  = lengthscale[0];
  float sc = -0.5f / (l * l);

  // Q A-fragments (DX = 64 -> two k=32 steps), hoisted; same in every wave
  const unsigned short* qrow = Qb + (size_t)(m * NQ_ + q0 + lo) * DX_;
  Frag aq0, aq1;
  aq0.h[0] = *(const u16x8*)(qrow + 16 * hi);
  aq0.h[1] = *(const u16x8*)(qrow + 16 * hi + 8);
  aq1.h[0] = *(const u16x8*)(qrow + 32 + 16 * hi);
  aq1.h[1] = *(const u16x8*)(qrow + 32 + 16 * hi + 8);

  // Phase 1: scores = sc * (||q||^2 + ||k||^2 - 2 q.k), masked, into LDS.
  // Wave w covers kv columns [128w, 128w+128).
  for (int kt = 0; kt < 8; ++kt) {
    int kv0 = w * 128 + kt * 16;
    const unsigned short* krow = Kb + (size_t)(m * NKV_ + kv0 + lo) * DX_;
    if (kt < 7)
      __builtin_prefetch(krow + 16 * DX_, 0, 1);   // next kv-tile row
    Frag b0, b1;
    b0.h[0] = *(const u16x8*)(krow + 16 * hi);
    b0.h[1] = *(const u16x8*)(krow + 16 * hi + 8);
    b1.h[0] = *(const u16x8*)(krow + 32 + 16 * hi);
    b1.h[1] = *(const u16x8*)(krow + 32 + 16 * hi + 8);
    v8f c = v8f{};
    c = wmma_bf16(aq0, b0, c);
    c = wmma_bf16(aq1, b1, c);
    for (int j = 0; j < 8; ++j) {
      int row = j + 8 * hi;                // local q row
      int q = q0 + row, kv = kv0 + lo;
      float s = sc * (qn[m * NQ_ + q] + kn[m * NKV_ + kv] - 2.0f * c[j]);
      if (!mask[((size_t)(m * NQ_) + q) * NKV_ + kv]) s = -3.4028235e38f;
      S[row * NKV_ + kv] = s;
    }
  }
  __syncthreads();

  // Phase 2: softmax. 128 lanes = 16 rows x 8 column-segments of 64.
  {
    int r   = tid & 15;
    int seg = tid >> 4;          // 0..7
    float* srow = S + r * NKV_ + seg * 64;
    float mx = -3.4028235e38f;
    for (int i = 0; i < 64; ++i) mx = fmaxf(mx, srow[i]);
    red[tid] = mx;
    __syncthreads();
    if (tid < 16) {
      float m2 = red[tid];
      for (int s = 1; s < 8; ++s) m2 = fmaxf(m2, red[tid + 16 * s]);
      rowmax[tid] = m2;
    }
    __syncthreads();
    float rm = rowmax[r];
    float sum = 0.f;
    unsigned short* prow = P + r * NKV_ + seg * 64;
    for (int i = 0; i < 64; ++i) {
      float e = __expf(srow[i] - rm);
      sum += e;
      prow[i] = f2bf(e);
    }
    red[tid] = sum;
    __syncthreads();
    if (tid < 16) {
      float s2 = 0.f;
      for (int s = 0; s < 8; ++s) s2 += red[tid + 16 * s];
      inv_sum[tid] = 1.0f / s2;
    }
  }
  __syncthreads();

  // Phase 3: O = P @ V (16 x 128). Wave w owns h-tiles {2w, 2w+1}.
  v8f acc[2];
  acc[0] = v8f{};
  acc[1] = v8f{};
  const unsigned short* prow  = P + lo * NKV_;
  const unsigned short* vbase = VbT + (size_t)m * DH_ * NKV_;
  for (int k0 = 0; k0 < NKV_; k0 += 32) {
    Frag a;
    a.h[0] = *(const u16x8*)(prow + k0 + 16 * hi);
    a.h[1] = *(const u16x8*)(prow + k0 + 16 * hi + 8);
    for (int tt = 0; tt < 2; ++tt) {
      int t = 2 * w + tt;
      const unsigned short* bp = vbase + (size_t)(16 * t + lo) * NKV_ + k0 + 16 * hi;
      Frag b;
      b.h[0] = *(const u16x8*)bp;
      b.h[1] = *(const u16x8*)(bp + 8);
      acc[tt] = wmma_bf16(a, b, acc[tt]);
    }
  }

  // Phase 4: scale by 1/rowsum, stage bf16 O row-major in LDS (layout fixup)
  for (int tt = 0; tt < 2; ++tt) {
    int t = 2 * w + tt;
    for (int j = 0; j < 8; ++j) {
      int row = j + 8 * hi;
      Ol[row * DH_ + 16 * t + lo] = f2bf(acc[tt][j] * inv_sum[row]);
    }
  }
  __syncthreads();

  // Phase 5: Y = O @ W_out + b (16 x 256). Wave w owns n-tiles {4w..4w+3}.
  Frag ao[4];
  const unsigned short* orow = Ol + lo * DH_;
  for (int kk = 0; kk < 4; ++kk) {
    ao[kk].h[0] = *(const u16x8*)(orow + kk * 32 + 16 * hi);
    ao[kk].h[1] = *(const u16x8*)(orow + kk * 32 + 16 * hi + 8);
  }
  for (int t2 = 4 * w; t2 < 4 * w + 4; ++t2) {
    int n0 = t2 * 16;
    const unsigned short* bp0 = WoT + (size_t)(n0 + lo) * DH_ + 16 * hi;
    v8f c = v8f{};
    for (int kk = 0; kk < 4; ++kk) {
      Frag b;
      b.h[0] = *(const u16x8*)(bp0 + kk * 32);
      b.h[1] = *(const u16x8*)(bp0 + kk * 32 + 8);
      c = wmma_bf16(ao[kk], b, c);
    }
    float bias = b_out[n0 + lo];
    for (int j = 0; j < 8; ++j) {
      int row = j + 8 * hi;
      out[((size_t)(m * NQ_) + q0 + row) * DY_ + n0 + lo] = c[j] + bias;
    }
  }
}

// ---------------- launch ----------------

extern "C" void kernel_launch(void* const* d_in, const int* in_sizes, int n_in,
                              void* d_out, int out_size, void* d_ws, size_t ws_size,
                              hipStream_t stream) {
  const float*         xq    = (const float*)d_in[0];
  const float*         xk    = (const float*)d_in[1];
  const float*         yv    = (const float*)d_in[2];
  const unsigned char* mask  = (const unsigned char*)d_in[3];
  const float*         W_v   = (const float*)d_in[4];
  const float*         W_out = (const float*)d_in[5];
  const float*         b_out = (const float*)d_in[6];
  const float*         ls    = (const float*)d_in[7];
  float*               out   = (float*)d_out;

  char* ws = (char*)d_ws;
  size_t off = 0;
  auto alloc = [&](size_t bytes) {
    char* p = ws + off;
    off += (bytes + 255) & ~(size_t)255;
    return p;
  };
  unsigned short* Qb  = (unsigned short*)alloc((size_t)M_ * NQ_  * DX_ * 2);
  unsigned short* Kb  = (unsigned short*)alloc((size_t)M_ * NKV_ * DX_ * 2);
  unsigned short* Yb  = (unsigned short*)alloc((size_t)M_ * NKV_ * DY_ * 2);
  unsigned short* WvT = (unsigned short*)alloc((size_t)DY_ * DH_ * 2);
  unsigned short* WoT = (unsigned short*)alloc((size_t)DH_ * DY_ * 2);
  unsigned short* VbT = (unsigned short*)alloc((size_t)M_ * DH_ * NKV_ * 2);
  float*          qn  = (float*)alloc((size_t)M_ * NQ_  * 4);
  float*          kn  = (float*)alloc((size_t)M_ * NKV_ * 4);

  int n;
  n = M_ * NQ_  * DX_; cvt_bf16<<<(n + 255) / 256, 256, 0, stream>>>(xq, Qb, n);
  n = M_ * NKV_ * DX_; cvt_bf16<<<(n + 255) / 256, 256, 0, stream>>>(xk, Kb, n);
  n = M_ * NKV_ * DY_; cvt_bf16<<<(n + 255) / 256, 256, 0, stream>>>(yv, Yb, n);
  n = DY_ * DH_; cvt_bf16_T<<<(n + 255) / 256, 256, 0, stream>>>(W_v,   WvT, DY_, DH_);
  n = DH_ * DY_; cvt_bf16_T<<<(n + 255) / 256, 256, 0, stream>>>(W_out, WoT, DH_, DY_);
  row_sumsq<<<(M_ * NQ_  + 255) / 256, 256, 0, stream>>>(xq, qn, M_ * NQ_,  DX_);
  row_sumsq<<<(M_ * NKV_ + 255) / 256, 256, 0, stream>>>(xk, kn, M_ * NKV_, DX_);

  vproj_kernel<<<M_ * (NKV_ / 16), 128, 0, stream>>>(Yb, WvT, VbT);
  attn_kernel<<<M_ * (NQ_ / 16), 128, 0, stream>>>(Qb, Kb, VbT, WoT, qn, kn,
                                                   mask, b_out, ls, out);
}
